// RNNLSTMModel_81724637708372
// MI455X (gfx1250) — matrix operand order repaired and checked by reading, not compiled, not run on hardware
//
#include <hip/hip_runtime.h>

#define VOCAB 32000
#define EMB   256
#define HID   256
#define BB    16
#define SS    256

typedef __attribute__((ext_vector_type(16))) __bf16 v16bf;
typedef __attribute__((ext_vector_type(8)))  float  v8f;

union BF16Frag { v16bf v; uint4 q[2]; };

__device__ __forceinline__ unsigned short f2bf(float f) {
    unsigned int u = __float_as_uint(f);
    u += 0x7FFFu + ((u >> 16) & 1u);   // round-to-nearest-even
    return (unsigned short)(u >> 16);
}

// fast sigmoid / tanh on the serial path: v_exp_f32 + v_rcp_f32
__device__ __forceinline__ float sigf(float x) {
    return __builtin_amdgcn_rcpf(1.0f + __expf(-x));
}
__device__ __forceinline__ float tanh_fast(float x) {
    float e = __expf(2.0f * x);
    return (e - 1.0f) * __builtin_amdgcn_rcpf(e + 1.0f);
}

// A-matrix fragment (16x32 bf16, MxK): lane half h holds K = k0+h*8..+7 and k0+16+h*8..+7
__device__ __forceinline__ v16bf load_a_frag(const unsigned short* X, int ld,
                                             int row_base, int k0, int lane) {
    int m = lane & 15, half = lane >> 4;
    const unsigned short* p = X + (row_base + m) * ld + k0 + half * 8;
    BF16Frag f;
    f.q[0] = *(const uint4*)(p);
    f.q[1] = *(const uint4*)(p + 16);
    return f.v;
}

// B-matrix fragment (32x16 bf16, KxN) from row-major W[N][K]: lane half h holds K = k0+h*8..+7 and k0+16+h*8..+7 of row n.
__device__ __forceinline__ v16bf load_b_frag(const unsigned short* W, int ld,
                                             int n_base, int k0, int lane) {
    int n = lane & 15, half = lane >> 4;
    const unsigned short* p = W + (n_base + n) * ld + k0 + half * 8;
    BF16Frag f;
    f.q[0] = *(const uint4*)(p);
    f.q[1] = *(const uint4*)(p + 16);
    return f.v;
}

#define WMMA_BF16(a, b, c) \
    __builtin_amdgcn_wmma_f32_16x16x32_bf16(false, (a), false, (b), (short)0, (c), false, false)

// ---------------- prep: f32 -> bf16 weight conversion -------------------------
__global__ void prep_bf16(const float* __restrict__ w_ih, const float* __restrict__ w_hh,
                          const float* __restrict__ w_dense,
                          unsigned short* __restrict__ wih_b, unsigned short* __restrict__ whh_b,
                          unsigned short* __restrict__ wd_b) {
    const int N_IH = 4 * HID * EMB;
    const int N_HH = 4 * HID * HID;
    const int N_D  = VOCAB * HID;
    const int total = N_IH + N_HH + N_D;
    for (int i = blockIdx.x * blockDim.x + threadIdx.x; i < total;
         i += gridDim.x * blockDim.x) {
        if (i < N_IH)                wih_b[i]               = f2bf(w_ih[i]);
        else if (i < N_IH + N_HH)    whh_b[i - N_IH]        = f2bf(w_hh[i - N_IH]);
        else                         wd_b[i - N_IH - N_HH]  = f2bf(w_dense[i - N_IH - N_HH]);
    }
}

// ---------------- embedding gather (time-major) + bf16 convert ----------------
__global__ void gather_x(const int* __restrict__ input, const float* __restrict__ emb,
                         unsigned short* __restrict__ x_b) {
    const int total = SS * BB * EMB;
    for (int i = blockIdx.x * blockDim.x + threadIdx.x; i < total;
         i += gridDim.x * blockDim.x) {
        int row = i >> 8;            // s*B + b
        int e   = i & 255;
        int s   = row >> 4;
        int b   = row & 15;
        int tok = input[b * SS + s];
        x_b[i] = f2bf(emb[tok * EMB + e]);
    }
}

// ---------------- x_gates = X @ w_ih^T + (b_ih + b_hh) ------------------------
// 4096 wave-tasks: mtile(=timestep) in [0,256), ngroup in [0,16) (4 N-tiles).
// Output stored FRAGMENT-MAJOR: XGT[t][ntile][lane][r] so the scan reads
// contiguous b128 pairs per tile. B loads use a 3-slot rotating buffer so two
// loads are always in flight ahead of the consuming WMMA.
__global__ __launch_bounds__(256) void gates_gemm(const unsigned short* __restrict__ Xb,
                                                  const unsigned short* __restrict__ Wb,
                                                  const float* __restrict__ b_ih,
                                                  const float* __restrict__ b_hh,
                                                  float* __restrict__ XGT) {
    int lane = threadIdx.x & 31;
    int wave = threadIdx.x >> 5;
    int task = blockIdx.x * 8 + wave;
    int mtile = task >> 4;
    int ngrp  = task & 15;
    int row_base = mtile * 16;
    int m = lane & 15;

    v16bf afr[8];
#pragma unroll
    for (int kc = 0; kc < 8; kc++)
        afr[kc] = load_a_frag(Xb, EMB, row_base, kc * 32, lane);

    v8f acc[4];
#pragma unroll
    for (int t = 0; t < 4; t++) {
        int n = (ngrp * 4 + t) * 16 + m;
        float bias = b_ih[n] + b_hh[n];
#pragma unroll
        for (int r = 0; r < 8; r++) acc[t][r] = bias;
    }

    v16bf bb[3];
    bb[0] = load_b_frag(Wb, EMB, (ngrp * 4 + 0) * 16, 0, lane);
    bb[1] = load_b_frag(Wb, EMB, (ngrp * 4 + 0) * 16, 32, lane);
#pragma unroll
    for (int i = 0; i < 32; i++) {           // i = t*8 + kc
        int t  = i >> 3;
        int kc = i & 7;
        if (i + 2 < 32)
            bb[(i + 2) % 3] = load_b_frag(Wb, EMB, (ngrp * 4 + ((i + 2) >> 3)) * 16,
                                          ((i + 2) & 7) * 32, lane);
        acc[t] = WMMA_BF16(afr[kc], bb[i % 3], acc[t]);
    }

#pragma unroll
    for (int t = 0; t < 4; t++) {
        int ntile = ngrp * 4 + t;
        // fragment-major: 8 consecutive floats per lane (32B store)
        *(v8f*)&XGT[((mtile * 64 + ntile) * 32 + lane) * 8] = acc[t];
    }
}

// ---------------- sequential LSTM scan: single workgroup, 16 waves ------------
// wave w owns hidden units [w*16, w*16+16): computes i/f/g/o tiles for those
// units (gate tiles w, w+16, w+32, w+48) so the c/h update is lane-local.
__global__ __launch_bounds__(512) void lstm_scan(const float* __restrict__ XGT,
                                                 const unsigned short* __restrict__ Whh,
                                                 const float* __restrict__ h0,
                                                 const float* __restrict__ c0,
                                                 unsigned short* __restrict__ Ys,
                                                 float* __restrict__ out_hc) {
    __shared__ __align__(16) unsigned short h_lds[BB * HID];

    int lane = threadIdx.x & 31;
    int w    = threadIdx.x >> 5;            // 0..15
    int mA   = lane & 15, halfA = lane >> 4;
    int m0   = halfA * 8;                   // 0 or 8
    int u    = w * 16 + mA;                 // hidden unit column

    v8f c, hreg;
#pragma unroll
    for (int r = 0; r < 8; r++) {
        int b = m0 + r;
        c[r]    = c0[b * HID + u];
        hreg[r] = h0[b * HID + u];
        h_lds[b * HID + u] = f2bf(hreg[r]);
    }
    __syncthreads();

    for (int t = 0; t < SS; t++) {
        // accumulator init: contiguous 32B fragment-major loads
        v8f acc[4];
#pragma unroll
        for (int g = 0; g < 4; g++) {
            int ntile = g * 16 + w;
            acc[g] = *(const v8f*)&XGT[((t * 64 + ntile) * 32 + lane) * 8];
        }
        // A fragments (h_{t-1}) from LDS, register-resident for the step
        v16bf afr[8];
#pragma unroll
        for (int kc = 0; kc < 8; kc++) {
            const uint4* pa = (const uint4*)&h_lds[mA * HID + kc * 32 + halfA * 8];
            BF16Frag fa;
            fa.q[0] = pa[0];
            fa.q[1] = pa[2];                 // +16 elements = +32 bytes
            afr[kc] = fa.v;
        }
        // g += h_{t-1} @ w_hh^T : 3-slot rotating B buffer from L2-hot w_hh
        v16bf bb[3];
        bb[0] = load_b_frag(Whh, HID, 0 * HID + w * 16, 0, lane);
        bb[1] = load_b_frag(Whh, HID, 0 * HID + w * 16, 32, lane);
#pragma unroll
        for (int i = 0; i < 32; i++) {       // i = g*8 + kc
            int g  = i >> 3;
            int kc = i & 7;
            if (i + 2 < 32)
                bb[(i + 2) % 3] = load_b_frag(Whh, HID, ((i + 2) >> 3) * HID + w * 16,
                                              ((i + 2) & 7) * 32, lane);
            acc[g] = WMMA_BF16(afr[kc], bb[i % 3], acc[g]);
        }
        __syncthreads();                     // all reads of h_{t-1} done
#pragma unroll
        for (int r = 0; r < 8; r++) {
            float ig = sigf(acc[0][r]);
            float fg = sigf(acc[1][r]);
            float gg = tanh_fast(acc[2][r]);
            float og = sigf(acc[3][r]);
            float cn = fg * c[r] + ig * gg;
            float hn = og * tanh_fast(cn);
            c[r]    = cn;
            hreg[r] = hn;
            unsigned short hb = f2bf(hn);
            h_lds[(m0 + r) * HID + u] = hb;
            Ys[(t * BB + m0 + r) * HID + u] = hb;
        }
        __syncthreads();                     // h_t visible to all waves
    }
    // final h (f32) then c (f32)
#pragma unroll
    for (int r = 0; r < 8; r++) {
        out_hc[(m0 + r) * HID + u]            = hreg[r];
        out_hc[BB * HID + (m0 + r) * HID + u] = c[r];
    }
}

// ---------------- dense head: out = ys @ w_dense^T + b_dense ------------------
// 64000 wave-tasks: mtile in [0,256), ngroup in [0,250) (8 N-tiles each).
// A fragments register-resident across 8 N-tiles; B double-buffered;
// write-once 524MB output stored non-temporal to keep w_dense/ys L2-resident.
__global__ __launch_bounds__(256) void dense_gemm(const unsigned short* __restrict__ Ysb,
                                                  const unsigned short* __restrict__ Wdb,
                                                  const float* __restrict__ b_dense,
                                                  float* __restrict__ Out) {
    int lane = threadIdx.x & 31;
    int wave = threadIdx.x >> 5;
    int task = blockIdx.x * 8 + wave;
    int mtile = task / 250;
    int ngrp  = task % 250;
    int row_base = mtile * 16;
    int m = lane & 15, half = lane >> 4;

    v16bf afr[8];
#pragma unroll
    for (int kc = 0; kc < 8; kc++)
        afr[kc] = load_a_frag(Ysb, HID, row_base, kc * 32, lane);

    v8f acc[8];
#pragma unroll
    for (int t = 0; t < 8; t++) {
        float bias = b_dense[(ngrp * 8 + t) * 16 + m];
#pragma unroll
        for (int r = 0; r < 8; r++) acc[t][r] = bias;
    }

    v16bf bcur = load_b_frag(Wdb, HID, (ngrp * 8 + 0) * 16, 0, lane);
#pragma unroll
    for (int i = 0; i < 64; i++) {           // i = t*8 + kc
        int t  = i >> 3;
        int kc = i & 7;
        v16bf bnext = bcur;
        if (i + 1 < 64)
            bnext = load_b_frag(Wdb, HID, (ngrp * 8 + ((i + 1) >> 3)) * 16,
                                ((i + 1) & 7) * 32, lane);
        acc[t] = WMMA_BF16(afr[kc], bcur, acc[t]);
        bcur = bnext;
    }

#pragma unroll
    for (int t = 0; t < 8; t++) {
        int n = (ngrp * 8 + t) * 16 + m;
#pragma unroll
        for (int r = 0; r < 8; r++)
            __builtin_nontemporal_store(acc[t][r],
                &Out[(row_base + half * 8 + r) * VOCAB + n]);
    }
}

// -----------------------------------------------------------------------------
extern "C" void kernel_launch(void* const* d_in, const int* in_sizes, int n_in,
                              void* d_out, int out_size, void* d_ws, size_t ws_size,
                              hipStream_t stream) {
    const int*   input   = (const int*)  d_in[0];
    const float* h0      = (const float*)d_in[1];
    const float* c0      = (const float*)d_in[2];
    const float* emb     = (const float*)d_in[3];
    const float* w_ih    = (const float*)d_in[4];
    const float* w_hh    = (const float*)d_in[5];
    const float* b_ih    = (const float*)d_in[6];
    const float* b_hh    = (const float*)d_in[7];
    const float* w_dense = (const float*)d_in[8];
    const float* b_dense = (const float*)d_in[9];
    float* out = (float*)d_out;

    // scratch layout (bytes, 256-aligned)
    char* ws = (char*)d_ws;
    unsigned short* x_b   = (unsigned short*)(ws + 0);                 //  2,097,152
    unsigned short* wih_b = (unsigned short*)(ws + 2097152);           //    524,288
    unsigned short* whh_b = (unsigned short*)(ws + 2621440);           //    524,288
    unsigned short* wd_b  = (unsigned short*)(ws + 3145728);           // 16,384,000
    float*          xgt   = (float*)        (ws + 19529728);           // 16,777,216
    unsigned short* ys_b  = (unsigned short*)(ws + 36306944);          //  2,097,152

    // 1) weight conversion + embedding gather
    prep_bf16<<<8192, 256, 0, stream>>>(w_ih, w_hh, w_dense, wih_b, whh_b, wd_b);
    gather_x<<<2048, 256, 0, stream>>>(input, emb, x_b);

    // 2) input projection: x_gates = X @ w_ih^T + (b_ih + b_hh), fragment-major
    gates_gemm<<<512, 256, 0, stream>>>(x_b, wih_b, b_ih, b_hh, xgt);

    // 3) sequential recurrence: one resident workgroup
    lstm_scan<<<1, 512, 0, stream>>>(xgt, whh_b, h0, c0, ys_b,
                                     out + (long long)SS * BB * VOCAB);

    // 4) dense head
    dense_gemm<<<8000, 256, 0, stream>>>(ys_b, wd_b, b_dense, out);
}
